// DownSampleCarve_43267500540730
// MI455X (gfx1250) — matrix operand order repaired and checked by reading, not compiled, not run on hardware
//
#include <hip/hip_runtime.h>
#include <hip/hip_bf16.h>
#include <math.h>

// ---------------- problem constants (match reference) ----------------
#define B_      8
#define C_      128
#define N_      2048
#define M_      512
#define K_      32
#define FFN_H_  512
#define SCALE_  0.08838834764831845f   // 1/sqrt(128)

typedef __bf16 bf16_t;
typedef __attribute__((ext_vector_type(16))) __bf16 v16bf;
typedef __attribute__((ext_vector_type(8)))  __bf16 v8bf;
typedef __attribute__((ext_vector_type(8)))  float  v8f;

#define WMMA_BF16(a, b, c) \
  __builtin_amdgcn_wmma_f32_16x16x32_bf16(false, (a), false, (b), (short)0, (c), false, false)

// CDNA5 WMMA VGPR layouts (ISA 7.12.2), wave32:
//   A 16x32 bf16 : lane&15 = row M, element e -> K = (e<8 ? e : e+8) + 8*(lane>>4)
//                  => per lane: two contiguous 8-element runs @ K = 8*hi and 16+8*hi
//   B 32x16 bf16 : lane&15 = col N, element e -> K = (e<8 ? e : e+8) + 8*(lane>>4)
//                  => per lane: two contiguous 8-element runs @ K = 8*hi and 16+8*hi
//   D 16x16 f32  : element e -> row M = e + 8*(lane>>4), col N = lane&15

// row = pointer to this lane's (row-major, K-contiguous) row; kbase = 32*ks
__device__ __forceinline__ v16bf load_a(const bf16_t* __restrict__ row, int kbase, int hi) {
  v8bf lo = *(const v8bf*)(row + kbase + 8 * hi);
  v8bf hv = *(const v8bf*)(row + kbase + 16 + 8 * hi);
  return __builtin_shufflevector(lo, hv, 0, 1, 2, 3, 4, 5, 6, 7,
                                 8, 9, 10, 11, 12, 13, 14, 15);
}
__device__ __forceinline__ v16bf load_b(const bf16_t* __restrict__ row, int kbase, int hi) {
  v8bf lo = *(const v8bf*)(row + kbase + 8 * hi);
  v8bf hv = *(const v8bf*)(row + kbase + 16 + 8 * hi);
  return __builtin_shufflevector(lo, hv, 0, 1, 2, 3, 4, 5, 6, 7,
                                 8, 9, 10, 11, 12, 13, 14, 15);
}

// =====================================================================
// 1) bin_conv: be = w1 @ x ; x2 = w2 @ [x;be]; keep x2 as bf16 (N,C),
//    f32 channel-0 row (only part ever reused), and per-point |x2|^2.
// =====================================================================
__global__ void k_x2(const float* __restrict__ x, const float* __restrict__ w1,
                     const float* __restrict__ w2, float* __restrict__ x2c0,
                     bf16_t* __restrict__ x2h_nc, float* __restrict__ sq) {
  __shared__ float xcol[C_];
  __shared__ float beL[2];
  __shared__ float red[C_];
  int bn = blockIdx.x;
  int b = bn / N_, n = bn % N_;
  int c = threadIdx.x;                       // 128 threads: one per out channel
  xcol[c] = x[(b * C_ + c) * N_ + n];
  __syncthreads();
  if (c < 2) {
    float acc = 0.f;
    for (int j = 0; j < C_; ++j) acc += w1[c * C_ + j] * xcol[j];
    beL[c] = acc;
  }
  __syncthreads();
  const float* wrow = w2 + c * (C_ + 2);     // bin_w2 is (C, C+2) row-major
  float acc = 0.f;
  for (int j = 0; j < C_; ++j) acc += wrow[j] * xcol[j];
  acc += wrow[C_] * beL[0] + wrow[C_ + 1] * beL[1];
  x2h_nc[(b * N_ + n) * C_ + c] = (bf16_t)acc;
  if (c == 0) x2c0[b * N_ + n] = acc;
  red[c] = acc * acc;
  __syncthreads();
  for (int s = 64; s > 0; s >>= 1) {
    if (c < s) red[c] += red[c + s];
    __syncthreads();
  }
  if (c == 0) sq[b * N_ + n] = red[0];
}

// =====================================================================
// 2) f32 -> bf16 weight conversion
// =====================================================================
__global__ void k_cvt_bf16(const float* __restrict__ src, bf16_t* __restrict__ dst, int n) {
  int i = blockIdx.x * blockDim.x + threadIdx.x;
  if (i < n) dst[i] = (bf16_t)src[i];
}

// =====================================================================
// 3) QKV: one wave per 16x16 output tile, K=128 -> 4 chained WMMA.
//    q,k stored (B,N,C) (K-contiguous for later B fragments);
//    v stored (B,C,N) (K=n contiguous for the attn@v^T GEMM).
// =====================================================================
__global__ void k_qkv(const bf16_t* __restrict__ wqh, const bf16_t* __restrict__ wkh,
                      const bf16_t* __restrict__ wvh, const bf16_t* __restrict__ x2h_nc,
                      bf16_t* __restrict__ qh, bf16_t* __restrict__ kh,
                      bf16_t* __restrict__ vh) {
  int lane = threadIdx.x & 31, widx = threadIdx.x >> 5;
  int hi = lane >> 4, l15 = lane & 15;
  int wg = blockIdx.x * 8 + widx;            // B*3*(C/16)*(N/16) waves
  int nt = wg % (N_ / 16); wg /= (N_ / 16);
  int ot = wg % (C_ / 16); wg /= (C_ / 16);
  int mat = wg % 3;
  int b = wg / 3;
  const bf16_t* W = (mat == 0) ? wqh : (mat == 1) ? wkh : wvh;
  const bf16_t* arow = W + (ot * 16 + l15) * C_;
  const bf16_t* brow = x2h_nc + (size_t)(b * N_ + nt * 16 + l15) * C_;
  v8f acc = {};
  for (int ks = 0; ks < C_ / 32; ++ks)
    acc = WMMA_BF16(load_a(arow, ks * 32, hi), load_b(brow, ks * 32, hi), acc);

  int n = nt * 16 + l15;
  if (mat < 2) {                             // q / k : (N,C), 8 contiguous -> 1 store
    v8bf ov;
#pragma unroll
    for (int e = 0; e < 8; ++e) ov[e] = (bf16_t)acc[e];
    bf16_t* dst = (mat == 0 ? qh : kh) + (size_t)(b * N_ + n) * C_ + ot * 16 + 8 * hi;
    *(v8bf*)dst = ov;
  } else {                                   // v : (C,N), strided scalar stores
#pragma unroll
    for (int e = 0; e < 8; ++e)
      vh[(size_t)(b * C_ + ot * 16 + e + 8 * hi) * N_ + n] = (bf16_t)acc[e];
  }
}

// =====================================================================
// 4) KNN: Gram via WMMA, distance rows in 128KB dynamic LDS, per-row
//    iterative top-32 smallest (ties -> lower index) with wave reductions.
// =====================================================================
__global__ void k_knn(const bf16_t* __restrict__ x2h_nc, const float* __restrict__ sq,
                      int* __restrict__ knn) {
  extern __shared__ float drow[];            // 16 * N_ floats
  int lane = threadIdx.x & 31, widx = threadIdx.x >> 5;   // 16 waves
  int hi = lane >> 4, l15 = lane & 15;
  int b = blockIdx.x / (N_ / 16);
  int ibase = (blockIdx.x % (N_ / 16)) * 16;

  const bf16_t* arow = x2h_nc + (size_t)(b * N_ + ibase + l15) * C_;
  v16bf afrag[4];
#pragma unroll
  for (int ks = 0; ks < 4; ++ks) afrag[ks] = load_a(arow, ks * 32, hi);

  for (int s = 0; s < 8; ++s) {
    int jt = widx + 16 * s;
    const bf16_t* brow = x2h_nc + (size_t)(b * N_ + jt * 16 + l15) * C_;
    v8f acc = {};
#pragma unroll
    for (int ks = 0; ks < 4; ++ks)
      acc = WMMA_BF16(afrag[ks], load_b(brow, ks * 32, hi), acc);
#pragma unroll
    for (int e = 0; e < 8; ++e) {
      int row = e + 8 * hi, j = jt * 16 + l15;
      drow[row * N_ + j] = sq[b * N_ + ibase + row] + sq[b * N_ + j] - 2.0f * acc[e];
    }
  }
  __syncthreads();
  int row = widx, r = ibase + row;           // one wave per row
  for (int t = 0; t < K_; ++t) {
    float bestv = 3.4e38f;
    int besti = 0x7fffffff;
    for (int j = lane; j < N_; j += 32) {
      float v = drow[row * N_ + j];
      if (v < bestv || (v == bestv && j < besti)) { bestv = v; besti = j; }
    }
#pragma unroll
    for (int mk = 16; mk > 0; mk >>= 1) {
      float ov = __shfl_xor(bestv, mk, 32);
      int   oi = __shfl_xor(besti, mk, 32);
      if (ov < bestv || (ov == bestv && oi < besti)) { bestv = ov; besti = oi; }
    }
    if (lane == 0) {
      knn[(b * N_ + r) * K_ + t] = besti;
      drow[row * N_ + besti] = 3.4e38f;      // in-wave DS ordering: safe
    }
  }
}

// =====================================================================
// 5) attention pass 1: energy rows in LDS, softmax per row, scatter the
//    32 knn entries of each row into aps numerator / count via atomics.
// =====================================================================
__global__ void k_pass1(const bf16_t* __restrict__ qh, const bf16_t* __restrict__ kh,
                        const int* __restrict__ knn, float* __restrict__ apsnum,
                        float* __restrict__ cnt) {
  extern __shared__ float erow[];            // 16 * N_ floats
  int lane = threadIdx.x & 31, widx = threadIdx.x >> 5;
  int hi = lane >> 4, l15 = lane & 15;
  int b = blockIdx.x / (N_ / 16);
  int ibase = (blockIdx.x % (N_ / 16)) * 16;

  const bf16_t* arow = qh + (size_t)(b * N_ + ibase + l15) * C_;
  v16bf afrag[4];
#pragma unroll
  for (int ks = 0; ks < 4; ++ks) afrag[ks] = load_a(arow, ks * 32, hi);

  for (int s = 0; s < 8; ++s) {
    int jt = widx + 16 * s;
    const bf16_t* brow = kh + (size_t)(b * N_ + jt * 16 + l15) * C_;
    v8f acc = {};
#pragma unroll
    for (int ks = 0; ks < 4; ++ks)
      acc = WMMA_BF16(afrag[ks], load_b(brow, ks * 32, hi), acc);
#pragma unroll
    for (int e = 0; e < 8; ++e)
      erow[(e + 8 * hi) * N_ + jt * 16 + l15] = acc[e];
  }
  __syncthreads();
  int row = widx, r = ibase + row;
  float m = -3.4e38f;
  for (int j = lane; j < N_; j += 32) m = fmaxf(m, erow[row * N_ + j]);
#pragma unroll
  for (int mk = 16; mk > 0; mk >>= 1) m = fmaxf(m, __shfl_xor(m, mk, 32));
  float s = 0.f;
  for (int j = lane; j < N_; j += 32) s += __expf((erow[row * N_ + j] - m) * SCALE_);
#pragma unroll
  for (int mk = 16; mk > 0; mk >>= 1) s += __shfl_xor(s, mk, 32);
  int jt = knn[(b * N_ + r) * K_ + lane];    // K_ == 32 == wave size
  float a = __expf((erow[row * N_ + jt] - m) * SCALE_) / s;
  atomicAdd(&apsnum[b * N_ + jt], a);
  atomicAdd(&cnt[b * N_ + jt], 1.0f);
}

__global__ void k_zero(float* __restrict__ p, int n) {
  int i = blockIdx.x * blockDim.x + threadIdx.x;
  if (i < n) p[i] = 0.f;
}

// =====================================================================
// 6) top-M: per-batch bitonic sort of 2048 (value desc, index asc)
// =====================================================================
__global__ void k_topm(const float* __restrict__ apsnum, const float* __restrict__ cnt,
                       int* __restrict__ idx_out, int* __restrict__ idx_ws) {
  extern __shared__ char sm[];
  float* val = (float*)sm;
  int* ind = (int*)(sm + N_ * sizeof(float));
  int b = blockIdx.x, tid = threadIdx.x;
  for (int i = tid; i < N_; i += 1024) {
    val[i] = apsnum[b * N_ + i] / (cnt[b * N_ + i] + 1e-8f);
    ind[i] = i;
  }
  __syncthreads();
  for (int k = 2; k <= N_; k <<= 1) {
    for (int j = k >> 1; j > 0; j >>= 1) {
      for (int t = tid; t < N_; t += 1024) {
        int ixj = t ^ j;
        if (ixj > t) {
          bool dir = ((t & k) == 0);
          float v1 = val[t], v2 = val[ixj];
          int i1 = ind[t], i2 = ind[ixj];
          bool before2 = (v2 > v1) || (v2 == v1 && i2 < i1);
          if (before2 == dir) { val[t] = v2; val[ixj] = v1; ind[t] = i2; ind[ixj] = i1; }
        }
      }
      __syncthreads();
    }
  }
  for (int m = tid; m < M_; m += 1024) {
    idx_out[b * M_ + m] = ind[m];
    idx_ws[b * M_ + m] = ind[m];
  }
}

// =====================================================================
// 7) pass 2 (flash-style): one wave per 16 selected rows. Recompute
//    energy (max pass + exp pass), route P through a per-wave LDS patch
//    (D-layout f32 -> A-layout bf16) straight into WMMA against v.
// =====================================================================
__global__ void k_pass2(const bf16_t* __restrict__ qh, const bf16_t* __restrict__ kh,
                        const bf16_t* __restrict__ vh, const int* __restrict__ idxb,
                        const float* __restrict__ x2c0, float* __restrict__ xds,
                        float* __restrict__ xtmp) {
  extern __shared__ float patch[];           // 8 waves * 16*32 floats
  int lane = threadIdx.x & 31, widx = threadIdx.x >> 5;
  int hi = lane >> 4, l15 = lane & 15;
  int wg = blockIdx.x * 8 + widx;            // B * M/16 waves
  int b = wg / (M_ / 16);
  int mblk = wg % (M_ / 16);
  int rA = idxb[b * M_ + mblk * 16 + l15];   // selected point for this A row
  float* pp = patch + widx * 16 * 32;

  const bf16_t* arow = qh + (size_t)(b * N_ + rA) * C_;
  v16bf afrag[4];
#pragma unroll
  for (int ks = 0; ks < 4; ++ks) afrag[ks] = load_a(arow, ks * 32, hi);

  // ---- pass A: row maxima ----
  float m8[8];
#pragma unroll
  for (int e = 0; e < 8; ++e) m8[e] = -3.4e38f;
  for (int jt = 0; jt < N_ / 16; ++jt) {
    const bf16_t* brow = kh + (size_t)(b * N_ + jt * 16 + l15) * C_;
    v8f acc = {};
#pragma unroll
    for (int ks = 0; ks < 4; ++ks)
      acc = WMMA_BF16(afrag[ks], load_b(brow, ks * 32, hi), acc);
#pragma unroll
    for (int e = 0; e < 8; ++e) m8[e] = fmaxf(m8[e], acc[e]);
  }
#pragma unroll
  for (int e = 0; e < 8; ++e)
#pragma unroll
    for (int mk = 1; mk < 16; mk <<= 1) m8[e] = fmaxf(m8[e], __shfl_xor(m8[e], mk, 32));

  // ---- pass B: P = exp((E-m)*scale); O += P_bf16 x v ----
  float s8[8];
#pragma unroll
  for (int e = 0; e < 8; ++e) s8[e] = 0.f;
  v8f oacc[8];
#pragma unroll
  for (int cb = 0; cb < 8; ++cb) oacc[cb] = (v8f){};
  for (int js = 0; js < N_ / 32; ++js) {
#pragma unroll
    for (int half = 0; half < 2; ++half) {
      int jt = js * 2 + half;
      const bf16_t* brow = kh + (size_t)(b * N_ + jt * 16 + l15) * C_;
      v8f acc = {};
#pragma unroll
      for (int ks = 0; ks < 4; ++ks)
        acc = WMMA_BF16(afrag[ks], load_b(brow, ks * 32, hi), acc);
#pragma unroll
      for (int e = 0; e < 8; ++e) {
        float p = __expf((acc[e] - m8[e]) * SCALE_);
        s8[e] += p;
        pp[(e + 8 * hi) * 32 + half * 16 + l15] = p;   // D-layout -> [row][col]
      }
    }
    // read back in A-layout: two contiguous 8-float runs per lane
    v8f plo = *(v8f*)(pp + l15 * 32 + 8 * hi);
    v8f phi = *(v8f*)(pp + l15 * 32 + 16 + 8 * hi);
    v16bf pa;
#pragma unroll
    for (int e = 0; e < 8; ++e) {
      pa[e]     = (bf16_t)plo[e];
      pa[e + 8] = (bf16_t)phi[e];
    }
    int j0 = js * 32;
#pragma unroll
    for (int cb = 0; cb < 8; ++cb) {
      const bf16_t* brow = vh + (size_t)(b * C_ + cb * 16 + l15) * N_;
      oacc[cb] = WMMA_BF16(pa, load_b(brow, j0, hi), oacc[cb]);
    }
  }
#pragma unroll
  for (int e = 0; e < 8; ++e)
#pragma unroll
    for (int mk = 1; mk < 16; mk <<= 1) s8[e] += __shfl_xor(s8[e], mk, 32);
#pragma unroll
  for (int cb = 0; cb < 8; ++cb)
#pragma unroll
    for (int e = 0; e < 8; ++e) {
      int c = cb * 16 + l15;
      int m = mblk * 16 + e + 8 * hi;
      xds[((b * C_) + c) * M_ + m] = oacc[cb][e] / s8[e];
    }
  if (hi == 0) xtmp[b * M_ + mblk * 16 + l15] = x2c0[b * N_ + rA];
}

// =====================================================================
// 8) BatchNorm stats over (B, M) per channel
// =====================================================================
__global__ void k_bnstats(const float* __restrict__ t, const float* __restrict__ addvec,
                          int use_add, float* __restrict__ mean_out,
                          float* __restrict__ istd_out) {
  __shared__ float s1[256], s2[256];
  int c = blockIdx.x, tid = threadIdx.x;
  float a = 0.f, q = 0.f;
  for (int i = tid; i < B_ * M_; i += 256) {
    int b = i / M_, m = i % M_;
    float v = t[((b * C_) + c) * M_ + m];
    if (use_add) v += addvec[b * M_ + m];
    a += v; q += v * v;
  }
  s1[tid] = a; s2[tid] = q;
  __syncthreads();
  for (int s = 128; s > 0; s >>= 1) {
    if (tid < s) { s1[tid] += s1[tid + s]; s2[tid] += s2[tid + s]; }
    __syncthreads();
  }
  if (tid == 0) {
    float mu = s1[0] / (float)(B_ * M_);
    float var = s2[0] / (float)(B_ * M_) - mu * mu;
    mean_out[c] = mu;
    istd_out[c] = rsqrtf(var + 1e-5f);
  }
}

// BN1 apply: reads xds (B,C,M) + xtmp, writes bf16 activations in (B,M,C)
__global__ void k_bn_apply_bf16(const float* __restrict__ t, const float* __restrict__ addvec,
                                const float* __restrict__ mean, const float* __restrict__ istd,
                                const float* __restrict__ g, const float* __restrict__ beta,
                                bf16_t* __restrict__ out) {
  int i = blockIdx.x * 256 + threadIdx.x;    // i indexes output (B,M,C)
  if (i >= B_ * C_ * M_) return;
  int c = i % C_, m = (i / C_) % M_, b = i / (C_ * M_);
  float v = t[((b * C_) + c) * M_ + m] + addvec[b * M_ + m];
  out[i] = (bf16_t)((v - mean[c]) * istd[c] * g[c] + beta[c]);
}

__global__ void k_bn_apply_f32(const float* __restrict__ t, const float* __restrict__ mean,
                               const float* __restrict__ istd, const float* __restrict__ g,
                               const float* __restrict__ beta, float* __restrict__ out) {
  int i = blockIdx.x * 256 + threadIdx.x;
  if (i >= B_ * C_ * M_) return;
  int c = (i / M_) % C_;
  out[i] = (t[i] - mean[c]) * istd[c] * g[c] + beta[c];
}

// =====================================================================
// 9) FFN: one wave per 16x16 tile, bf16 WMMA, fully vectorized fragments
//    xr1 in (B,M,C); h in (B,M,FFN_H)
// =====================================================================
__global__ void k_ffn1(const bf16_t* __restrict__ w1h, const bf16_t* __restrict__ xr1h,
                       bf16_t* __restrict__ h) {
  int lane = threadIdx.x & 31, widx = threadIdx.x >> 5;
  int hi = lane >> 4, l15 = lane & 15;
  int wg = blockIdx.x * 8 + widx;            // B*(FFN_H/16)*(M/16)
  int mt = wg % (M_ / 16); wg /= (M_ / 16);
  int ot = wg % (FFN_H_ / 16);
  int b = wg / (FFN_H_ / 16);
  const bf16_t* arow = w1h + (ot * 16 + l15) * C_;
  const bf16_t* brow = xr1h + (size_t)(b * M_ + mt * 16 + l15) * C_;
  v8f acc = {};
  for (int ks = 0; ks < C_ / 32; ++ks)
    acc = WMMA_BF16(load_a(arow, ks * 32, hi), load_b(brow, ks * 32, hi), acc);
  v8bf ov;
#pragma unroll
  for (int e = 0; e < 8; ++e) {
    float v = acc[e];
    ov[e] = (bf16_t)(v > 0.f ? v : 0.2f * v);   // LeakyReLU(0.2)
  }
  *(v8bf*)(h + (size_t)(b * M_ + mt * 16 + l15) * FFN_H_ + ot * 16 + 8 * hi) = ov;
}

__global__ void k_ffn2(const bf16_t* __restrict__ w2h, const bf16_t* __restrict__ h,
                       const float* __restrict__ xds, float* __restrict__ t2) {
  int lane = threadIdx.x & 31, widx = threadIdx.x >> 5;
  int hi = lane >> 4, l15 = lane & 15;
  int wg = blockIdx.x * 8 + widx;            // B*(C/16)*(M/16)
  int mt = wg % (M_ / 16); wg /= (M_ / 16);
  int ct = wg % (C_ / 16);
  int b = wg / (C_ / 16);
  const bf16_t* arow = w2h + (ct * 16 + l15) * FFN_H_;
  const bf16_t* brow = h + (size_t)(b * M_ + mt * 16 + l15) * FFN_H_;
  v8f acc = {};
  for (int ks = 0; ks < FFN_H_ / 32; ++ks)
    acc = WMMA_BF16(load_a(arow, ks * 32, hi), load_b(brow, ks * 32, hi), acc);
#pragma unroll
  for (int e = 0; e < 8; ++e) {
    int o = ((b * C_) + ct * 16 + e + 8 * hi) * M_ + mt * 16 + l15;
    t2[o] = acc[e] + xds[o];
  }
}

// =====================================================================
// host side
// =====================================================================
static inline char* carve(char*& p, size_t bytes) {
  char* r = p;
  p += (bytes + 255) & ~(size_t)255;
  return r;
}

extern "C" void kernel_launch(void* const* d_in, const int* in_sizes, int n_in,
                              void* d_out, int out_size, void* d_ws, size_t ws_size,
                              hipStream_t stream) {
  const float* x      = (const float*)d_in[0];
  const float* bin_w1 = (const float*)d_in[1];
  const float* bin_w2 = (const float*)d_in[2];
  const float* wq     = (const float*)d_in[3];
  const float* wk     = (const float*)d_in[4];
  const float* wv     = (const float*)d_in[5];
  const float* bn1_g  = (const float*)d_in[6];
  const float* bn1_b  = (const float*)d_in[7];
  const float* ffn_w1 = (const float*)d_in[8];
  const float* ffn_w2 = (const float*)d_in[9];
  const float* bn2_g  = (const float*)d_in[10];
  const float* bn2_b  = (const float*)d_in[11];

  char* p = (char*)d_ws;
  float*  x2c0   = (float*) carve(p, (size_t)B_ * N_ * 4);
  bf16_t* x2h_nc = (bf16_t*)carve(p, (size_t)B_ * N_ * C_ * 2);
  float*  sq     = (float*) carve(p, (size_t)B_ * N_ * 4);
  bf16_t* wqh    = (bf16_t*)carve(p, (size_t)C_ * C_ * 2);
  bf16_t* wkh    = (bf16_t*)carve(p, (size_t)C_ * C_ * 2);
  bf16_t* wvh    = (bf16_t*)carve(p, (size_t)C_ * C_ * 2);
  bf16_t* w1h    = (bf16_t*)carve(p, (size_t)FFN_H_ * C_ * 2);
  bf16_t* w2h    = (bf16_t*)carve(p, (size_t)C_ * FFN_H_ * 2);
  bf16_t* qh     = (bf16_t*)carve(p, (size_t)B_ * N_ * C_ * 2);
  bf16_t* kh     = (bf16_t*)carve(p, (size_t)B_ * N_ * C_ * 2);
  bf16_t* vh     = (bf16_t*)carve(p, (size_t)B_ * C_ * N_ * 2);
  int*    knn    = (int*)   carve(p, (size_t)B_ * N_ * K_ * 4);
  float*  apsnum = (float*) carve(p, (size_t)B_ * N_ * 4);
  float*  cntb   = (float*) carve(p, (size_t)B_ * N_ * 4);
  int*    idxb   = (int*)   carve(p, (size_t)B_ * M_ * 4);
  float*  xds    = (float*) carve(p, (size_t)B_ * C_ * M_ * 4);
  float*  xtmp   = (float*) carve(p, (size_t)B_ * M_ * 4);
  float*  mean1  = (float*) carve(p, C_ * 4);
  float*  istd1  = (float*) carve(p, C_ * 4);
  float*  mean2  = (float*) carve(p, C_ * 4);
  float*  istd2  = (float*) carve(p, C_ * 4);
  bf16_t* xr1h   = (bf16_t*)carve(p, (size_t)B_ * M_ * C_ * 2);
  bf16_t* hbuf   = (bf16_t*)carve(p, (size_t)B_ * M_ * FFN_H_ * 2);
  float*  t2     = (float*) carve(p, (size_t)B_ * C_ * M_ * 4);

  float* xres_out = (float*)d_out;                        // (B,C,M)
  int*   idx_out  = (int*)d_out + (size_t)B_ * C_ * M_;   // (B,1,M) int32

  // 1) x2 / be / sq / bf16 copy
  k_x2<<<B_ * N_, C_, 0, stream>>>(x, bin_w1, bin_w2, x2c0, x2h_nc, sq);

  // 2) weight conversions
  k_cvt_bf16<<<(C_ * C_ + 255) / 256, 256, 0, stream>>>(wq, wqh, C_ * C_);
  k_cvt_bf16<<<(C_ * C_ + 255) / 256, 256, 0, stream>>>(wk, wkh, C_ * C_);
  k_cvt_bf16<<<(C_ * C_ + 255) / 256, 256, 0, stream>>>(wv, wvh, C_ * C_);
  k_cvt_bf16<<<(FFN_H_ * C_ + 255) / 256, 256, 0, stream>>>(ffn_w1, w1h, FFN_H_ * C_);
  k_cvt_bf16<<<(C_ * FFN_H_ + 255) / 256, 256, 0, stream>>>(ffn_w2, w2h, C_ * FFN_H_);

  // 3) q, k, v
  k_qkv<<<(B_ * 3 * (C_ / 16) * (N_ / 16)) / 8, 256, 0, stream>>>(wqh, wkh, wvh, x2h_nc,
                                                                  qh, kh, vh);

  // 4) knn (128KB dynamic LDS per WG)
  k_knn<<<B_ * (N_ / 16), 512, 16 * N_ * sizeof(float), stream>>>(x2h_nc, sq, knn);

  // 5) aps accumulation
  k_zero<<<(B_ * N_ + 255) / 256, 256, 0, stream>>>(apsnum, B_ * N_);
  k_zero<<<(B_ * N_ + 255) / 256, 256, 0, stream>>>(cntb, B_ * N_);
  k_pass1<<<B_ * (N_ / 16), 512, 16 * N_ * sizeof(float), stream>>>(qh, kh, knn,
                                                                    apsnum, cntb);

  // 6) top-M selection (writes idx output too)
  k_topm<<<B_, 1024, N_ * (sizeof(float) + sizeof(int)), stream>>>(apsnum, cntb,
                                                                   idx_out, idxb);

  // 7) attn_down @ v, flash-style
  k_pass2<<<(B_ * (M_ / 16)) / 8, 256, 8 * 16 * 32 * sizeof(float), stream>>>(
      qh, kh, vh, idxb, x2c0, xds, xtmp);

  // 8) BN1 -> FFN -> BN2
  k_bnstats<<<C_, 256, 0, stream>>>(xds, xtmp, 1, mean1, istd1);
  k_bn_apply_bf16<<<(B_ * C_ * M_ + 255) / 256, 256, 0, stream>>>(xds, xtmp, mean1, istd1,
                                                                  bn1_g, bn1_b, xr1h);
  k_ffn1<<<(B_ * (FFN_H_ / 16) * (M_ / 16)) / 8, 256, 0, stream>>>(w1h, xr1h, hbuf);
  k_ffn2<<<(B_ * (C_ / 16) * (M_ / 16)) / 8, 256, 0, stream>>>(w2h, hbuf, xds, t2);
  k_bnstats<<<C_, 256, 0, stream>>>(t2, xtmp, 0, mean2, istd2);
  k_bn_apply_f32<<<(B_ * C_ * M_ + 255) / 256, 256, 0, stream>>>(t2, mean2, istd2,
                                                                 bn2_g, bn2_b, xres_out);
  (void)in_sizes; (void)n_in; (void)out_size; (void)ws_size;
}
